// BEVFormerEncoderLayer_10677288698575
// MI455X (gfx1250) — compile-verified, matrix operand order checked by reading, not run on hardware
//
#include <hip/hip_runtime.h>
#include <hip/hip_bf16.h>

// ---------------- problem constants ----------------
#define QN   2500
#define CN   256
#define NHN  8
#define HDN  32
#define VN   6
#define SUMHW 34016
// level table: (H,W,row-offset)
// (100,256,0) (50,128,25600) (25,64,32000) (13,32,33600)

typedef __attribute__((ext_vector_type(16))) __bf16 v16bf;
typedef __attribute__((ext_vector_type(8)))  float  v8f;

static __device__ __forceinline__ __bf16 f2bf(float x) { return (__bf16)x; }

// =====================================================================
// Generic GEMM  D[M,N] = A[M,K] @ W[K,N] + bias  (bf16 WMMA, f32 acc)
// block = 256 threads = 8 waves; wave grid 2x4 -> 32x64 tile per block
// Weight tile staged to LDS via async global->LDS (ASYNCcnt) each K-chunk.
// OUT_BF16: 0 -> f32 output, 1 -> bf16 output
// =====================================================================
#define LDS_STRIDE 68   // 64 cols + pad; 68*4 = 272 B (16B multiple, bank-shifted)

template<int OUT_BF16>
__global__ __launch_bounds__(256)
void gemm_wmma_bf16(const float* __restrict__ A, const float* __restrict__ W,
                    const float* __restrict__ bias, void* __restrict__ Dst,
                    int M, int N, int K, int relu)
{
    __shared__ float lb[32 * LDS_STRIDE];   // staged 32(K) x 64(N) f32 tile

    const int lane = threadIdx.x & 31;
    const int wave = threadIdx.x >> 5;
    const int hsel = lane >> 4;        // which K-half this lane holds
    const int l16  = lane & 15;
    const int m0  = blockIdx.y * 32 + (wave >> 2) * 16;
    const int n0b = blockIdx.x * 64;
    const int n0  = n0b + (wave & 3) * 16;

    int am = m0 + l16; if (am > M - 1) am = M - 1;   // clamp ragged M edge
    const float* __restrict__ arow = A + (size_t)am * K;
    const int bn = n0 + l16;                          // output column (may be >= N)
    const int bcol_t = (wave & 3) * 16 + l16;         // column within 64-wide tile

    const uint32_t lds_base = (uint32_t)(uintptr_t)(&lb[0]);

    v8f acc = {0.f, 0.f, 0.f, 0.f, 0.f, 0.f, 0.f, 0.f};

    for (int k0 = 0; k0 < K; k0 += 32) {
        // ---- async-stage the 32x64 f32 weight tile into LDS (ASYNCcnt path)
#pragma unroll
        for (int i = 0; i < 2; ++i) {
            const int ch = (int)threadIdx.x + i * 256;   // 512 chunks of 4 floats
            const int kk = ch >> 4;
            const int nn = (ch & 15) * 4;
            int nsrc = n0b + nn; if (nsrc > N - 4) nsrc = N - 4;   // clamp narrow N
            const unsigned long long ga =
                (unsigned long long)(uintptr_t)(W + (size_t)(k0 + kk) * N + nsrc);
            const uint32_t la = lds_base + (uint32_t)((kk * LDS_STRIDE + nn) * 4);
            asm volatile("global_load_async_to_lds_b128 %0, %1, off"
                         :: "v"(la), "v"(ga) : "memory");
        }
        asm volatile("s_wait_asynccnt 0x0" ::: "memory");
        __syncthreads();

        const int ka = k0 + hsel * 8;
        if (k0 + 32 < K) __builtin_prefetch(arow + ka + 32, 0, 1);

        // A fragment: contiguous global, lane holds K = ka..ka+7 and ka+16..ka+23
        v16bf af, bfv;
#pragma unroll
        for (int j = 0; j < 8; ++j) af[j]     = f2bf(arow[ka + j]);
#pragma unroll
        for (int j = 0; j < 8; ++j) af[8 + j] = f2bf(arow[ka + 16 + j]);
        // B fragment from LDS tile (k rows hsel*8+j and hsel*8+16+j, column bcol_t)
#pragma unroll
        for (int j = 0; j < 8; ++j)
            bfv[j]     = f2bf(lb[(hsel * 8 + j) * LDS_STRIDE + bcol_t]);
#pragma unroll
        for (int j = 0; j < 8; ++j)
            bfv[8 + j] = f2bf(lb[(hsel * 8 + j + 16) * LDS_STRIDE + bcol_t]);

        acc = __builtin_amdgcn_wmma_f32_16x16x32_bf16(
                  false, af, false, bfv, (short)0, acc, false, false);

        __syncthreads();   // tile reused next iteration
    }

    const float bv = bias ? bias[bn < N ? bn : N - 1] : 0.0f;
#pragma unroll
    for (int r = 0; r < 8; ++r) {
        const int m = m0 + hsel * 8 + r;   // C/D layout: VGPR r, lanes>=16 hold M+8
        if (m < M && bn < N) {
            float o = acc[r] + bv;
            if (relu) o = fmaxf(o, 0.0f);
            if (OUT_BF16) ((__bf16*)Dst)[(size_t)m * N + bn] = (__bf16)o;
            else          ((float*)Dst)[(size_t)m * N + bn]  = o;
        }
    }
}

// fold concat([q,q]) @ Woff(512,64) -> q @ (Woff[0:256]+Woff[256:512])
__global__ void combine_woff(const float* __restrict__ Woff, float* __restrict__ out)
{
    int i = blockIdx.x * blockDim.x + threadIdx.x;
    if (i < 256 * 64) out[i] = Woff[i] + Woff[i + 256 * 64];
}

// softmax over last G elements, one thread per row (G <= 64)
__global__ void softmax_inplace(float* __restrict__ x, int rows, int G)
{
    int r = blockIdx.x * blockDim.x + threadIdx.x;
    if (r >= rows) return;
    float* p = x + (size_t)r * G;
    float mx = -3.0e38f;
    for (int i = 0; i < G; ++i) mx = fmaxf(mx, p[i]);
    float s = 0.f;
    for (int i = 0; i < G; ++i) { float e = __expf(p[i] - mx); p[i] = e; s += e; }
    float inv = 1.0f / s;
    for (int i = 0; i < G; ++i) p[i] *= inv;
}

// ---------------- TSA MSDA: one wave per (q,h), lane = head-dim channel ----
__global__ __launch_bounds__(256)
void tsa_msda(const float* __restrict__ val,   // (2500, 8, 32), grid order y*50+x
              const float* __restrict__ ref,   // (2500, 2)
              const float* __restrict__ off,   // (2500, 64) = (Q, NH*NP*2)
              const float* __restrict__ aw,    // (2500, 32) = (Q, NH*NP)
              float* __restrict__ out)         // (2500, 256)
{
    int unit = blockIdx.x * 8 + (threadIdx.x >> 5);
    if (unit >= QN * NHN) return;
    int q = unit >> 3, h = unit & 7, lane = threadIdx.x & 31;
    const float rx = ref[q * 2 + 0], ry = ref[q * 2 + 1];
    float acc = 0.f;
#pragma unroll
    for (int p = 0; p < 4; ++p) {
        const float ox = off[(size_t)q * 64 + h * 8 + p * 2 + 0];
        const float oy = off[(size_t)q * 64 + h * 8 + p * 2 + 1];
        const float x = (rx + ox * 0.02f) * 50.0f - 0.5f;   // /BEV_W then *W
        const float y = (ry + oy * 0.02f) * 50.0f - 0.5f;
        const float w = aw[(size_t)q * 32 + h * 4 + p];
        const float x0f = floorf(x), y0f = floorf(y);
        const int   x0 = (int)x0f,  y0 = (int)y0f;
        const float fx = x - x0f,   fy = y - y0f;
        float g = 0.f;
#pragma unroll
        for (int dy = 0; dy < 2; ++dy) {
#pragma unroll
            for (int dx = 0; dx < 2; ++dx) {
                const int xi = x0 + dx, yi = y0 + dy;
                if (xi >= 0 && xi < 50 && yi >= 0 && yi < 50) {
                    const float cw = (dx ? fx : 1.f - fx) * (dy ? fy : 1.f - fy);
                    g += cw * val[((size_t)(yi * 50 + xi) * 8 + h) * 32 + lane];
                }
            }
        }
        acc += w * g;
    }
    out[(size_t)q * 256 + h * 32 + lane] = acc;
}

// ---------------- SCA MSDA over 6 views x 4 levels, bf16 value tensor -----
__global__ __launch_bounds__(256)
void sca_msda(const void* __restrict__ svalv,          // (6, 34016, 8, 32) bf16
              const float* __restrict__ ref,           // (2500, 6, 4, 4, 2)
              const float* __restrict__ off,           // (2500, 1024)
              const float* __restrict__ aw,            // (2500, 512)
              const unsigned char* __restrict__ mask,  // (2500, 6) bool bytes
              float* __restrict__ out)                 // (2500, 256)
{
    const __bf16* __restrict__ sval = (const __bf16*)svalv;
    int unit = blockIdx.x * 8 + (threadIdx.x >> 5);
    if (unit >= QN * NHN) return;
    int q = unit >> 3, h = unit & 7, lane = threadIdx.x & 31;

    float acc = 0.f, cnt = 0.f;
    for (int v = 0; v < VN; ++v) {
        const __bf16* __restrict__ vb = sval + (size_t)v * SUMHW * 256;
        float ov = 0.f;
#pragma unroll
        for (int l = 0; l < 4; ++l) {
            const int   H  = (l == 0 ? 100 : l == 1 ? 50 : l == 2 ? 25 : 13);
            const int   W  = (l == 0 ? 256 : l == 1 ? 128 : l == 2 ? 64 : 32);
            const int   s0 = (l == 0 ? 0 : l == 1 ? 25600 : l == 2 ? 32000 : 33600);
            const float invW = 1.0f / (float)W, invH = 1.0f / (float)H;
            const __bf16* __restrict__ lg = vb + (size_t)s0 * 256;
            for (int z = 0; z < 4; ++z) {
                const float rx = ref[(size_t)(q * 6 + v) * 32 + l * 8 + z * 2 + 0];
                const float ry = ref[(size_t)(q * 6 + v) * 32 + l * 8 + z * 2 + 1];
                for (int p = 0; p < 4; ++p) {
                    const float ox = off[(size_t)q * 1024 + h * 128 + l * 32 + z * 8 + p * 2 + 0];
                    const float oy = off[(size_t)q * 1024 + h * 128 + l * 32 + z * 8 + p * 2 + 1];
                    const float x = (rx + ox * invW) * (float)W - 0.5f;
                    const float y = (ry + oy * invH) * (float)H - 0.5f;
                    const float w = aw[(size_t)q * 512 + h * 64 + l * 16 + z * 4 + p];
                    const float x0f = floorf(x), y0f = floorf(y);
                    const int   x0 = (int)x0f,  y0 = (int)y0f;
                    const float fx = x - x0f,   fy = y - y0f;
                    float g = 0.f;
#pragma unroll
                    for (int dy = 0; dy < 2; ++dy) {
#pragma unroll
                        for (int dx = 0; dx < 2; ++dx) {
                            const int xi = x0 + dx, yi = y0 + dy;
                            if (xi >= 0 && xi < W && yi >= 0 && yi < H) {
                                const float cw = (dx ? fx : 1.f - fx) * (dy ? fy : 1.f - fy);
                                g += cw * (float)lg[((size_t)(yi * W + xi) * 8 + h) * 32 + lane];
                            }
                        }
                    }
                    ov += w * g;
                }
            }
        }
        const float mv = mask[q * 6 + v] ? 1.f : 0.f;
        acc += ov * mv;
        cnt += mv;
    }
    out[(size_t)q * 256 + h * 32 + lane] = acc / fmaxf(cnt, 1.f);
}

// ---------------- residual + LayerNorm(256), one wave per row -------------
__global__ __launch_bounds__(256)
void residual_ln(const float* __restrict__ x, const float* __restrict__ res,
                 const float* __restrict__ gg, const float* __restrict__ bb,
                 float* __restrict__ out, int rows)
{
    int wave = threadIdx.x >> 5, lane = threadIdx.x & 31;
    int row = blockIdx.x * 8 + wave;
    if (row >= rows) return;
    const float* __restrict__ xp = x + (size_t)row * 256;
    const float* __restrict__ rp = res + (size_t)row * 256;
    float v[8]; float s = 0.f;
#pragma unroll
    for (int i = 0; i < 8; ++i) { v[i] = xp[lane + 32 * i] + rp[lane + 32 * i]; s += v[i]; }
#pragma unroll
    for (int o = 16; o > 0; o >>= 1) s += __shfl_xor(s, o, 32);
    const float mean = s * (1.0f / 256.0f);
    float vs = 0.f;
#pragma unroll
    for (int i = 0; i < 8; ++i) { float d = v[i] - mean; vs += d * d; }
#pragma unroll
    for (int o = 16; o > 0; o >>= 1) vs += __shfl_xor(vs, o, 32);
    const float rstd = rsqrtf(vs * (1.0f / 256.0f) + 1e-5f);
#pragma unroll
    for (int i = 0; i < 8; ++i)
        out[(size_t)row * 256 + lane + 32 * i] =
            (v[i] - mean) * rstd * gg[lane + 32 * i] + bb[lane + 32 * i];
}

// =====================================================================
static void launch_gemm(const float* A, const float* W, const float* b, void* D,
                        int M, int N, int K, int relu, int out_bf16, hipStream_t s)
{
    dim3 grid((N + 63) / 64, (M + 31) / 32);
    if (out_bf16) gemm_wmma_bf16<1><<<grid, 256, 0, s>>>(A, W, b, D, M, N, K, relu);
    else          gemm_wmma_bf16<0><<<grid, 256, 0, s>>>(A, W, b, D, M, N, K, relu);
}

extern "C" void kernel_launch(void* const* d_in, const int* in_sizes, int n_in,
                              void* d_out, int out_size, void* d_ws, size_t ws_size,
                              hipStream_t stream)
{
    (void)in_sizes; (void)n_in; (void)out_size; (void)ws_size;
    const float* q     = (const float*)d_in[0];
    const float* fmaps = (const float*)d_in[2];
    const float* sref  = (const float*)d_in[3];
    const unsigned char* smask = (const unsigned char*)d_in[4];
    const float* tref  = (const float*)d_in[7];
    const float* tsa_Wv   = (const float*)d_in[8];
    const float* tsa_bv   = (const float*)d_in[9];
    const float* tsa_Woff = (const float*)d_in[10];
    const float* tsa_boff = (const float*)d_in[11];
    const float* tsa_Wa   = (const float*)d_in[12];
    const float* tsa_ba   = (const float*)d_in[13];
    const float* tsa_Wo   = (const float*)d_in[14];
    const float* tsa_bo   = (const float*)d_in[15];
    const float* sca_Wv   = (const float*)d_in[16];
    const float* sca_bv   = (const float*)d_in[17];
    const float* sca_Woff = (const float*)d_in[18];
    const float* sca_boff = (const float*)d_in[19];
    const float* sca_Wa   = (const float*)d_in[20];
    const float* sca_ba   = (const float*)d_in[21];
    const float* sca_Wo   = (const float*)d_in[22];
    const float* sca_bo   = (const float*)d_in[23];
    const float* ln1_g = (const float*)d_in[24];
    const float* ln1_b = (const float*)d_in[25];
    const float* ln2_g = (const float*)d_in[26];
    const float* ln2_b = (const float*)d_in[27];
    const float* ln3_g = (const float*)d_in[28];
    const float* ln3_b = (const float*)d_in[29];
    const float* mlp_W1 = (const float*)d_in[30];
    const float* mlp_b1 = (const float*)d_in[31];
    const float* mlp_W2 = (const float*)d_in[32];
    const float* mlp_b2 = (const float*)d_in[33];

    // workspace carve-up (all chunks are multiples of 16 floats -> 64B aligned)
    float* ws    = (float*)d_ws;
    float* val   = ws;                   // 640000
    float* off1  = val   + 640000;       // 160000
    float* aw1   = off1  + 160000;       // 80000
    float* msda1 = aw1   + 80000;        // 640000
    float* out1  = msda1 + 640000;       // 640000
    float* out2  = out1  + 640000;       // 640000
    float* off2  = out2  + 640000;       // 2560000
    float* aw2   = off2  + 2560000;      // 1280000
    float* msda2 = aw2   + 1280000;      // 640000
    float* out3  = msda2 + 640000;       // 640000
    float* out4  = out3  + 640000;       // 640000
    float* hid   = out4  + 640000;       // 1280000
    float* out5  = hid   + 1280000;      // 640000
    float* wcomb = out5  + 640000;       // 16384
    void*  sval  = (void*)(wcomb + 16384); // 52,248,576 bf16 (~104 MB)

    // ---------------- TSA ----------------
    combine_woff<<<64, 256, 0, stream>>>(tsa_Woff, wcomb);
    launch_gemm(q, tsa_Wv, tsa_bv, val,  QN, 256,  256, 0, 0, stream);
    launch_gemm(q, wcomb,  tsa_boff, off1, QN, 64, 256, 0, 0, stream);
    launch_gemm(q, tsa_Wa, tsa_ba, aw1,  QN, 32,  256, 0, 0, stream);
    softmax_inplace<<<(QN * NHN + 255) / 256, 256, 0, stream>>>(aw1, QN * NHN, 4);
    tsa_msda<<<QN, 256, 0, stream>>>(val, tref, off1, aw1, msda1);
    launch_gemm(msda1, tsa_Wo, tsa_bo, out1, QN, 256, 256, 0, 0, stream);
    residual_ln<<<(QN + 7) / 8, 256, 0, stream>>>(out1, q, ln1_g, ln1_b, out2, QN);

    // ---------------- SCA ----------------
    launch_gemm(fmaps, sca_Wv, sca_bv, sval, VN * SUMHW, 256, 256, 0, 1, stream);
    launch_gemm(q, sca_Woff, sca_boff, off2, QN, 1024, 256, 0, 0, stream);
    launch_gemm(q, sca_Wa,   sca_ba,   aw2,  QN, 512,  256, 0, 0, stream);
    softmax_inplace<<<(QN * NHN + 255) / 256, 256, 0, stream>>>(aw2, QN * NHN, 64);
    sca_msda<<<QN, 256, 0, stream>>>(sval, sref, off2, aw2, smask, msda2);
    launch_gemm(msda2, sca_Wo, sca_bo, out3, QN, 256, 256, 0, 0, stream);
    residual_ln<<<(QN + 7) / 8, 256, 0, stream>>>(out3, out2, ln2_g, ln2_b, out4, QN);

    // ---------------- FFN ----------------
    launch_gemm(out4, mlp_W1, mlp_b1, hid,  QN, 512, 256, 1, 0, stream);
    launch_gemm(hid,  mlp_W2, mlp_b2, out5, QN, 256, 512, 0, 0, stream);
    residual_ln<<<(QN + 7) / 8, 256, 0, stream>>>(out5, out4, ln3_g, ln3_b, (float*)d_out, QN);
}
